// MultiHeadAttention_33277406609852
// MI455X (gfx1250) — compile-verified, matrix operand order checked by reading, not run on hardware
//
#include <hip/hip_runtime.h>
#include <hip/hip_bf16.h>
#include <cstdint>

// ---------------------------------------------------------------------------
// MI455X (gfx1250): wave32, WMMA f32_16x16x32_f16 matrix cores,
// Tensor Data Mover (tensor_load_to_lds) for global->LDS tile staging.
// ---------------------------------------------------------------------------

typedef _Float16 v16h __attribute__((ext_vector_type(16)));
typedef _Float16 v8h  __attribute__((ext_vector_type(8)));
typedef float    v8f  __attribute__((ext_vector_type(8)));
typedef unsigned int u32x4 __attribute__((ext_vector_type(4)));
typedef int          i32x4 __attribute__((ext_vector_type(4)));
typedef int          i32x8 __attribute__((ext_vector_type(8)));

#define B_    64
#define L_    512
#define D_    320          // 256 + 64
#define H_    8
#define HD_   64           // KS == VS == 64
#define OUT_  256
#define NTOK  (B_ * L_)    // 32768
#define NQKV  1536         // 3 * H*KS

// ----------------------------- WMMA helpers --------------------------------

__device__ __forceinline__ v8f wmma_f16(v16h a, v16h b, v8f c) {
  return __builtin_amdgcn_wmma_f32_16x16x32_f16(false, a, false, b, (short)0, c,
                                                false, false);
}

// A-fragment (16x32 f16) from LDS tile [rows][ld] row-major.
// lanes 0-15: rows m, halves 0..7 = K 0..7, 8..15 = K 16..23
// lanes 16-31: rows m, halves 0..7 = K 8..15, 8..15 = K 24..31
__device__ __forceinline__ v16h load_frag_a(const _Float16* lds, int ld,
                                            int rowbase, int kofs) {
  int lane = threadIdx.x & 31;
  int m    = lane & 15;
  int kb   = (lane < 16) ? 0 : 8;
  const _Float16* p = lds + (rowbase + m) * ld + kofs + kb;
  v8h lo = *(const v8h*)(p);
  v8h hi = *(const v8h*)(p + 16);
  v16h r;
#pragma unroll
  for (int j = 0; j < 8; ++j) { r[j] = lo[j]; r[j + 8] = hi[j]; }
  return r;
}

// B-fragment (32x16 f16) from LDS tile stored [K][ld] row-major (lane = col).
__device__ __forceinline__ v16h load_frag_b(const _Float16* lds, int ld,
                                            int colbase, int kofs) {
  int lane = threadIdx.x & 31;
  int n    = colbase + (lane & 15);
  int kb   = (lane < 16) ? 0 : 8;
  v16h r;
#pragma unroll
  for (int j = 0; j < 8; ++j) {
    r[j]     = lds[(kofs + kb + j) * ld + n];
    r[j + 8] = lds[(kofs + kb + 16 + j) * ld + n];
  }
  return r;
}

// --------------------- Tensor Data Mover 2D tile load ----------------------
// DMA a [rows x width] f16 tile (row stride = stride_elems) from global
// memory into LDS at lds_off. D# per CDNA5 ISA ch.8 (CDNA5 ISA).
// This toolchain exposes the 6-arg builtin:
//   (u32x4 g0, i32x8 g1, i32x4 g2, i32x4 g3, i32x8 pad, i32 cpol)

__device__ __forceinline__ void tdm_load_2d(unsigned int lds_off,
                                            const void* gptr,
                                            unsigned int width,
                                            unsigned int rows,
                                            unsigned int stride_elems) {
  unsigned long long ga = (unsigned long long)(uintptr_t)gptr;
  u32x4 g0;
  g0[0] = 1u;                                    // count=1, user flags 0
  g0[1] = lds_off;                               // lds_addr [63:32]
  g0[2] = (unsigned int)(ga & 0xffffffffu);      // global_addr [95:64]
  g0[3] = (unsigned int)((ga >> 32) & 0x01ffffffu) | (2u << 30);  // type=2
  i32x8 g1;
  g1[0] = (int)(1u << 16);                       // data_size=1 (2 bytes)
  g1[1] = (int)((width & 0xffffu) << 16);        // tensor_dim0[15:0]
  g1[2] = (int)((width >> 16) | ((rows & 0xffffu) << 16));  // td0 hi | td1 lo
  g1[3] = (int)(((rows >> 16) & 0xffffu) | (width << 16));  // td1 hi | tile_dim0
  g1[4] = (int)(rows & 0xffffu);                 // tile_dim1 (tile_dim2=0)
  g1[5] = (int)stride_elems;                     // tensor_dim0_stride[31:0]
  g1[6] = 0;                                     // stride hi / dim1_stride lo
  g1[7] = 0;
  i32x4 z4 = {0, 0, 0, 0};
  i32x8 z8 = {0, 0, 0, 0, 0, 0, 0, 0};
  __builtin_amdgcn_tensor_load_to_lds(g0, g1, z4, z4, z8, 0);
}

__device__ __forceinline__ unsigned int lds_offset_of(const void* p) {
  return (unsigned int)(uintptr_t)p;  // LDS aperture: addr[31:0] == LDS offset
}

// ----------------------------- pack kernels --------------------------------

__global__ void pack_x_kernel(const float* __restrict__ emb,
                              const float* __restrict__ boxes,
                              _Float16* __restrict__ X16) {
  int idx = blockIdx.x * 256 + threadIdx.x;
  if (idx >= NTOK * D_) return;
  int row = idx / D_, c = idx % D_;
  float v = (c < 256) ? emb[(size_t)row * 256 + c]
                      : boxes[(size_t)row * 64 + (c - 256)];
  X16[idx] = (_Float16)v;
}

__global__ void pack_wqkv_kernel(const float* __restrict__ Wq,
                                 const float* __restrict__ Wk,
                                 const float* __restrict__ Wv,
                                 _Float16* __restrict__ W) {
  int idx = blockIdx.x * 256 + threadIdx.x;
  if (idx >= D_ * NQKV) return;
  int d = idx / NQKV, n = idx % NQKV;
  float v = (n < 512)    ? Wq[(size_t)d * 512 + n]
            : (n < 1024) ? Wk[(size_t)d * 512 + n - 512]
                         : Wv[(size_t)d * 512 + n - 1024];
  W[idx] = (_Float16)v;
}

__global__ void cvt_f16_kernel(const float* __restrict__ src,
                               _Float16* __restrict__ dst, int n) {
  int idx = blockIdx.x * 256 + threadIdx.x;
  if (idx < n) dst[idx] = (_Float16)src[idx];
}

// ------------------------------ WMMA GEMM ----------------------------------
// C[M,N] = A[M,K](f16) * B[K,N](f16) + bias.
// 128x64 tile per 128-thread block (4 waves); wave w owns rows [32w,32w+32).
// Tiles staged by the Tensor Data Mover: wave0 DMAs the A tile, wave1 the B
// tile; TENSORcnt + workgroup barrier publish the LDS tiles.

template <bool F16OUT>
__global__ __launch_bounds__(128) void gemm_wmma_kernel(
    const _Float16* __restrict__ A, const _Float16* __restrict__ Bm,
    const float* __restrict__ bias, void* __restrict__ Cout,
    int M, int N, int K) {
  __shared__ __align__(16) _Float16 sA[128 * 32];
  __shared__ __align__(16) _Float16 sB[32 * 64];
  const int n0 = blockIdx.x * 64, m0 = blockIdx.y * 128;
  const int tid = threadIdx.x, w = tid >> 5, lane = tid & 31;
  const int hl = lane >> 4, nlo = lane & 15;

  v8f acc[2][4] = {};
  for (int k0 = 0; k0 < K; k0 += 32) {
    if (w == 0) {
      tdm_load_2d(lds_offset_of(sA), &A[(size_t)m0 * K + k0], 32, 128, K);
      __builtin_amdgcn_s_wait_tensorcnt(0);
    } else if (w == 1) {
      tdm_load_2d(lds_offset_of(sB), &Bm[(size_t)k0 * N + n0], 64, 32, N);
      __builtin_amdgcn_s_wait_tensorcnt(0);
    }
    __syncthreads();

    v16h bf[4];
#pragma unroll
    for (int cg = 0; cg < 4; ++cg) bf[cg] = load_frag_b(sB, 64, cg * 16, 0);
#pragma unroll
    for (int sr = 0; sr < 2; ++sr) {
      v16h af = load_frag_a(sA, 32, w * 32 + sr * 16, 0);
#pragma unroll
      for (int cg = 0; cg < 4; ++cg)
        acc[sr][cg] = wmma_f16(af, bf[cg], acc[sr][cg]);
    }
    __syncthreads();
  }

#pragma unroll
  for (int sr = 0; sr < 2; ++sr)
#pragma unroll
    for (int cg = 0; cg < 4; ++cg)
#pragma unroll
      for (int r = 0; r < 8; ++r) {
        int row = m0 + w * 32 + sr * 16 + r + 8 * hl;
        int col = n0 + cg * 16 + nlo;
        float v = acc[sr][cg][r] + (bias ? bias[col] : 0.f);
        if constexpr (F16OUT)
          ((_Float16*)Cout)[(size_t)row * N + col] = (_Float16)v;
        else
          ((float*)Cout)[(size_t)row * N + col] = v;
      }
}

// --------------------------- flash attention -------------------------------
// One block = (b, h, 64-row q tile); 4 waves, each owns a 16-row strip.
// Q/K/V tiles DMA-staged by the TDM.

__global__ __launch_bounds__(128) void attn_kernel(
    const _Float16* __restrict__ qkv,   // [NTOK][1536] f16 (Q|K|V)
    const int* __restrict__ lengths,
    _Float16* __restrict__ Z) {         // [NTOK][512] f16
  __shared__ __align__(16) _Float16 sQ[64 * 64];
  __shared__ __align__(16) _Float16 sK[64 * 64];
  __shared__ __align__(16) _Float16 sV[64 * 64];
  __shared__ __align__(16) _Float16 sP[64 * 64];

  const int q0 = blockIdx.x * 64, h = blockIdx.y, b = blockIdx.z;
  const int tid = threadIdx.x, w = tid >> 5, lane = tid & 31;
  const int hl = lane >> 4, nlo = lane & 15;
  const int len = lengths[b];
  const size_t rowbase = (size_t)b * L_;

  // stage Q tile (row-major [64 rows][64 d]) via TDM
  if (w == 0) {
    tdm_load_2d(lds_offset_of(sQ), &qkv[(rowbase + q0) * NQKV + h * 64],
                64, 64, NQKV);
    __builtin_amdgcn_s_wait_tensorcnt(0);
  }

  float mrow[8], lrow[8];
  v8f o[4] = {};
#pragma unroll
  for (int r = 0; r < 8; ++r) { mrow[r] = -1e30f; lrow[r] = 0.f; }

  for (int kb = 0; kb < 8; ++kb) {
    // stage K and V blocks via TDM (protected by previous end-of-iter barrier)
    if (w == 0) {
      tdm_load_2d(lds_offset_of(sK),
                  &qkv[(rowbase + kb * 64) * NQKV + 512 + h * 64], 64, 64,
                  NQKV);
      __builtin_amdgcn_s_wait_tensorcnt(0);
    } else if (w == 1) {
      tdm_load_2d(lds_offset_of(sV),
                  &qkv[(rowbase + kb * 64) * NQKV + 1024 + h * 64], 64, 64,
                  NQKV);
      __builtin_amdgcn_s_wait_tensorcnt(0);
    }
    __syncthreads();

    // S = Q K^T : Kt B-fragment == A-style contiguous read of sK[key][d]
    v8f s[4] = {};
#pragma unroll
    for (int ks = 0; ks < 64; ks += 32) {
      v16h aq = load_frag_a(sQ, 64, w * 16, ks);
#pragma unroll
      for (int cg = 0; cg < 4; ++cg) {
        v16h bk = load_frag_a(sK, 64, cg * 16, ks);
        s[cg] = wmma_f16(aq, bk, s[cg]);
      }
    }

    // scale + key mask
#pragma unroll
    for (int cg = 0; cg < 4; ++cg) {
      int key = kb * 64 + cg * 16 + nlo;
      bool ok = key < len;
#pragma unroll
      for (int r = 0; r < 8; ++r) {
        float v = s[cg][r] * 0.125f;        // 1/sqrt(64)
        s[cg][r] = ok ? v : -1e30f;
      }
    }

    // online softmax: rows live in 16-lane halves (lane0-15: r, lane16-31: r+8)
#pragma unroll
    for (int r = 0; r < 8; ++r) {
      float mx = s[0][r];
#pragma unroll
      for (int cg = 1; cg < 4; ++cg) mx = fmaxf(mx, s[cg][r]);
#pragma unroll
      for (int d = 1; d < 16; d <<= 1) mx = fmaxf(mx, __shfl_xor(mx, d));
      float mnew = fmaxf(mrow[r], mx);
      float f = __expf(mrow[r] - mnew);
      mrow[r] = mnew;
      float sum = 0.f;
#pragma unroll
      for (int cg = 0; cg < 4; ++cg) {
        float p = __expf(s[cg][r] - mnew);
        s[cg][r] = p;
        sum += p;
      }
#pragma unroll
      for (int d = 1; d < 16; d <<= 1) sum += __shfl_xor(sum, d);
      lrow[r] = lrow[r] * f + sum;
#pragma unroll
      for (int cg = 0; cg < 4; ++cg) o[cg][r] *= f;
    }

    // P -> LDS in A-layout source form (row-major [64 rows][64 keys])
#pragma unroll
    for (int cg = 0; cg < 4; ++cg)
#pragma unroll
      for (int r = 0; r < 8; ++r)
        sP[(w * 16 + r + 8 * hl) * 64 + cg * 16 + nlo] = (_Float16)s[cg][r];
    __syncthreads();

    // O += P V
#pragma unroll
    for (int ks = 0; ks < 64; ks += 32) {
      v16h ap = load_frag_a(sP, 64, w * 16, ks);
#pragma unroll
      for (int cg = 0; cg < 4; ++cg) {
        v16h bv = load_frag_b(sV, 64, cg * 16, ks);
        o[cg] = wmma_f16(ap, bv, o[cg]);
      }
    }
    __syncthreads();
  }

  // normalize and write Z (f16) at [token][h*64 + v]
#pragma unroll
  for (int r = 0; r < 8; ++r) {
    float inv = 1.f / lrow[r];
    int row = q0 + w * 16 + r + 8 * hl;
#pragma unroll
    for (int cg = 0; cg < 4; ++cg)
      Z[(rowbase + row) * 512 + h * 64 + cg * 16 + nlo] =
          (_Float16)(o[cg][r] * inv);
  }
}

// -------------------------- masked BatchNorm -------------------------------

__global__ void zero_stats_kernel(float* __restrict__ stats) {
  int i = blockIdx.x * 256 + threadIdx.x;
  if (i < 512) stats[i] = 0.f;
}

__global__ void bn_stats_kernel(const float* __restrict__ X,
                                const int* __restrict__ lengths,
                                float* __restrict__ stats) {
  int c = threadIdx.x;
  float s = 0.f, ss = 0.f;
  for (int row = blockIdx.x; row < NTOK; row += gridDim.x) {
    int b = row >> 9, pos = row & 511;
    if (pos < lengths[b]) {
      float v = X[(size_t)row * OUT_ + c];
      s += v;
      ss += v * v;
    }
  }
  atomicAdd(&stats[c], s);
  atomicAdd(&stats[OUT_ + c], ss);
}

__global__ void bn_apply1_kernel(const float* __restrict__ Z1,
                                 const float* __restrict__ stats,
                                 const float* __restrict__ g,
                                 const float* __restrict__ be,
                                 const int* __restrict__ lengths,
                                 _Float16* __restrict__ out) {
  __shared__ float cntS;
  int row = blockIdx.x, c = threadIdx.x;
  if (c == 0) {
    int t = 0;
    for (int i = 0; i < B_; ++i) t += lengths[i];
    cntS = (float)t;
  }
  __syncthreads();
  float cnt = cntS;
  float mean = stats[c] / cnt;
  float var = stats[OUT_ + c] / cnt - mean * mean;
  float inv = rsqrtf(var + 1e-5f);
  int b = row >> 9, pos = row & 511;
  float msk = (pos < lengths[b]) ? 1.f : 0.f;
  float x = Z1[(size_t)row * OUT_ + c];
  float y = g[c] * (x - mean) * inv + be[c];
  out[(size_t)row * OUT_ + c] = (_Float16)(fmaxf(y, 0.f) * msk);
}

__global__ void bn_apply2_kernel(const float* __restrict__ Z2,
                                 const float* __restrict__ stats,
                                 const float* __restrict__ g,
                                 const float* __restrict__ be,
                                 const float* __restrict__ emb,
                                 const int* __restrict__ lengths,
                                 float* __restrict__ out) {
  __shared__ float cntS;
  int row = blockIdx.x, c = threadIdx.x;
  if (c == 0) {
    int t = 0;
    for (int i = 0; i < B_; ++i) t += lengths[i];
    cntS = (float)t;
  }
  __syncthreads();
  float cnt = cntS;
  float mean = stats[c] / cnt;
  float var = stats[OUT_ + c] / cnt - mean * mean;
  float inv = rsqrtf(var + 1e-5f);
  int b = row >> 9, pos = row & 511;
  float msk = (pos < lengths[b]) ? 1.f : 0.f;
  float x = Z2[(size_t)row * OUT_ + c];
  float y = g[c] * (x - mean) * inv + be[c] + emb[(size_t)row * OUT_ + c];
  out[(size_t)row * OUT_ + c] = fmaxf(y, 0.f) * msk;
}

// ------------------------------- launcher ----------------------------------

extern "C" void kernel_launch(void* const* d_in, const int* in_sizes, int n_in,
                              void* d_out, int out_size, void* d_ws,
                              size_t ws_size, hipStream_t stream) {
  const float* emb    = (const float*)d_in[0];
  const float* boxes  = (const float*)d_in[1];
  const int*   lens   = (const int*)d_in[2];
  const float* Wq     = (const float*)d_in[3];
  const float* Wk     = (const float*)d_in[4];
  const float* Wv     = (const float*)d_in[5];
  const float* W1     = (const float*)d_in[6];
  const float* b1     = (const float*)d_in[7];
  const float* W2     = (const float*)d_in[8];
  const float* b2     = (const float*)d_in[9];
  const float* g1     = (const float*)d_in[10];
  const float* be1    = (const float*)d_in[11];
  const float* g2     = (const float*)d_in[12];
  const float* be2    = (const float*)d_in[13];
  float* out = (float*)d_out;

  char* ws = (char*)d_ws;
  size_t o = 0;
  _Float16* X16  = (_Float16*)(ws + o); o += (size_t)NTOK * D_ * 2;
  _Float16* WQKV = (_Float16*)(ws + o); o += (size_t)D_ * NQKV * 2;
  _Float16* W116 = (_Float16*)(ws + o); o += (size_t)512 * OUT_ * 2;
  _Float16* W216 = (_Float16*)(ws + o); o += (size_t)OUT_ * OUT_ * 2;
  _Float16* QKV  = (_Float16*)(ws + o); o += (size_t)NTOK * NQKV * 2;
  _Float16* Z16  = (_Float16*)(ws + o); o += (size_t)NTOK * 512 * 2;
  float*    Z1F  = (float*)(ws + o);    o += (size_t)NTOK * OUT_ * 4;
  _Float16* Z1H  = (_Float16*)(ws + o); o += (size_t)NTOK * OUT_ * 2;
  float*    Z2F  = (float*)(ws + o);    o += (size_t)NTOK * OUT_ * 4;
  float*    STAT = (float*)(ws + o);    o += 512 * 4;

  // 1. pack inputs / weights to f16
  pack_x_kernel<<<(NTOK * D_ + 255) / 256, 256, 0, stream>>>(emb, boxes, X16);
  pack_wqkv_kernel<<<(D_ * NQKV + 255) / 256, 256, 0, stream>>>(Wq, Wk, Wv, WQKV);
  cvt_f16_kernel<<<(512 * OUT_ + 255) / 256, 256, 0, stream>>>(W1, W116, 512 * OUT_);
  cvt_f16_kernel<<<(OUT_ * OUT_ + 255) / 256, 256, 0, stream>>>(W2, W216, OUT_ * OUT_);

  // 2. QKV projection: [32768,320] x [320,1536] -> f16
  gemm_wmma_kernel<true><<<dim3(NQKV / 64, NTOK / 128), 128, 0, stream>>>(
      X16, WQKV, nullptr, QKV, NTOK, NQKV, D_);

  // 3. attention (flash, per (b,h,qtile))
  attn_kernel<<<dim3(L_ / 64, H_, B_), 128, 0, stream>>>(QKV, lens, Z16);

  // 4. Z @ W1 + b1 -> f32
  gemm_wmma_kernel<false><<<dim3(OUT_ / 64, NTOK / 128), 128, 0, stream>>>(
      Z16, W116, b1, Z1F, NTOK, OUT_, 512);

  // 5. masked BN #1 -> ReLU -> mask (f16)
  zero_stats_kernel<<<2, 256, 0, stream>>>(STAT);
  bn_stats_kernel<<<256, 256, 0, stream>>>(Z1F, lens, STAT);
  bn_apply1_kernel<<<NTOK, OUT_, 0, stream>>>(Z1F, STAT, g1, be1, lens, Z1H);

  // 6. Z1 @ W2 + b2 -> f32
  gemm_wmma_kernel<false><<<dim3(OUT_ / 64, NTOK / 128), 128, 0, stream>>>(
      Z1H, W216, b2, Z2F, NTOK, OUT_, OUT_);

  // 7. masked BN #2 + skip(emb) -> ReLU -> mask (f32 out)
  zero_stats_kernel<<<2, 256, 0, stream>>>(STAT);
  bn_stats_kernel<<<256, 256, 0, stream>>>(Z2F, lens, STAT);
  bn_apply2_kernel<<<NTOK, OUT_, 0, stream>>>(Z2F, STAT, g2, be2, emb, lens, out);
}
